// moded_TimesSeriesAttention_39874476376349
// MI455X (gfx1250) — matrix operand order repaired and checked by reading, not compiled, not run on hardware
//
#include <hip/hip_runtime.h>
#include <hip/hip_bf16.h>

// Shapes (fixed by the reference): B=8, T=64, F=16, E=256, H=8, d=32.
// M = B*T*F = 131072 rows.

typedef __attribute__((ext_vector_type(16))) _Float16 v16h;
typedef __attribute__((ext_vector_type(8)))  _Float16 v8h;
typedef __attribute__((ext_vector_type(8)))  float    v8f;
typedef __attribute__((ext_vector_type(4)))  float    v4f;
typedef __attribute__((ext_vector_type(2)))  float    v2f;

union V16 { v16h v; v8h h2[2]; };

#define WMMA_F16(A, B, C) \
  __builtin_amdgcn_wmma_f32_16x16x32_f16(false, (A), false, (B), (short)0, (C), false, false)

#define EDIM 256
#define MTOT 131072
#define LDSW 40          // padded n-major LDS row stride (halves): conflict-free b128 reads

static __device__ __forceinline__ v8f vzero8() {
  v8f z;
  #pragma unroll
  for (int i = 0; i < 8; ++i) z[i] = 0.f;
  return z;
}

// ---------------------------------------------------------------------------
// Kernel 1: Y_f16 = f16( X_f32 @ W_f32 + bias ).  One WG = 128 rows x all 256 cols.
// 8 waves, each wave owns a 16-row strip and 16 accumulator tiles (16x16 each).
// ---------------------------------------------------------------------------
__global__ __launch_bounds__(256) void proj_f16_kernel(
    const float* __restrict__ X, const float* __restrict__ W,
    const float* __restrict__ bias, _Float16* __restrict__ Yh)
{
  __shared__ _Float16 Wt[EDIM * LDSW];   // 20 KB: 32-k chunk of W, n-major, f16
  const int tid   = threadIdx.x;
  const int wave  = tid >> 5;
  const int lane  = tid & 31;
  const int nlo   = lane & 15;
  const int hf    = lane >> 4;
  const int mbase = blockIdx.x * 128 + wave * 16;
  const int mrow  = mbase + nlo;

  v8f acc[16];
  #pragma unroll
  for (int t = 0; t < 16; ++t) acc[t] = vzero8();

  for (int k0 = 0; k0 < EDIM; k0 += 32) {
    __syncthreads();
    // Stage W[k0..k0+32) transposed into LDS as f16 (coalesced float4 reads).
    #pragma unroll
    for (int it = 0; it < 8; ++it) {
      const int i  = tid + it * 256;
      const int kk = i >> 6;            // 0..31
      const int n4 = (i & 63) << 2;     // 0..252
      const v4f w4 = *(const v4f*)(W + (size_t)(k0 + kk) * EDIM + n4);
      Wt[(n4 + 0) * LDSW + kk] = (_Float16)w4.x;
      Wt[(n4 + 1) * LDSW + kk] = (_Float16)w4.y;
      Wt[(n4 + 2) * LDSW + kk] = (_Float16)w4.z;
      Wt[(n4 + 3) * LDSW + kk] = (_Float16)w4.w;
    }
    __syncthreads();

    // A fragment (16x32 f16): per-lane row mrow; k = k0 + {hf*8+j, 16+hf*8+j}.
    const float* ap = X + (size_t)mrow * EDIM + k0 + hf * 8;
    const v4f a0 = *(const v4f*)(ap);
    const v4f a1 = *(const v4f*)(ap + 4);
    const v4f a2 = *(const v4f*)(ap + 16);
    const v4f a3 = *(const v4f*)(ap + 20);
    V16 af;
    #pragma unroll
    for (int j = 0; j < 4; ++j) {
      af.v[j]      = (_Float16)a0[j];
      af.v[j + 4]  = (_Float16)a1[j];
      af.v[j + 8]  = (_Float16)a2[j];
      af.v[j + 12] = (_Float16)a3[j];
    }
    // 16 B tiles from LDS: per-lane col n, k = k0 + {hf*8+j, 16+hf*8+j}.
    #pragma unroll
    for (int nt = 0; nt < 16; ++nt) {
      const _Float16* bp = &Wt[(nt * 16 + nlo) * LDSW + hf * 8];
      V16 bf;
      bf.h2[0] = *(const v8h*)(bp);
      bf.h2[1] = *(const v8h*)(bp + 16);
      acc[nt] = WMMA_F16(af.v, bf.v, acc[nt]);
    }
  }

  // Epilogue: + bias, convert, store f16. C-frag: lane col = nlo, rows r + 8*hf.
  #pragma unroll
  for (int nt = 0; nt < 16; ++nt) {
    const int n = nt * 16 + nlo;
    const float bb = bias[n];
    #pragma unroll
    for (int r = 0; r < 8; ++r) {
      const int m = mbase + r + hf * 8;
      Yh[(size_t)m * EDIM + n] = (_Float16)(acc[nt][r] + bb);
    }
  }
}

// ---------------------------------------------------------------------------
// Kernel 2a: per-query softmax stats over keys.  A = K-tile (keys as rows),
// B = Q-tile (queries as cols, loaded once).  Online max/sum across 64 key tiles.
// ---------------------------------------------------------------------------
__global__ __launch_bounds__(256) void softmax_stats_kernel(
    const _Float16* __restrict__ Qh, const _Float16* __restrict__ Kh,
    v2f* __restrict__ stats)
{
  const int bh = blockIdx.x;
  const int b = bh >> 3, h = bh & 7;
  const int wave = threadIdx.x >> 5, lane = threadIdx.x & 31;
  const int nlo = lane & 15, hf = lane >> 4;
  const int qt = blockIdx.y * 8 + wave;            // query tile 0..63
  const float scale = 0.17677669529663687f;        // 1/sqrt(32)

  const size_t qrow = ((size_t)b * 1024 + qt * 16 + nlo) * EDIM + h * 32;
  V16 bf;
  bf.h2[0] = *(const v8h*)(Qh + qrow + hf * 8);
  bf.h2[1] = *(const v8h*)(Qh + qrow + 16 + hf * 8);

  float m = -1e30f, Z = 0.f;
  for (int kt = 0; kt < 64; ++kt) {
    const size_t krow = ((size_t)b * 1024 + kt * 16 + nlo) * EDIM + h * 32;
    V16 af;
    af.h2[0] = *(const v8h*)(Kh + krow + hf * 8);
    af.h2[1] = *(const v8h*)(Kh + krow + 16 + hf * 8);
    v8f c = vzero8();
    c = WMMA_F16(af.v, bf.v, c);
    float tm = c[0] * scale;
    #pragma unroll
    for (int r = 1; r < 8; ++r) tm = fmaxf(tm, c[r] * scale);
    const float mn = fmaxf(m, tm);
    float zs = 0.f;
    #pragma unroll
    for (int r = 0; r < 8; ++r) zs += __expf(c[r] * scale - mn);
    Z = Z * __expf(m - mn) + zs;
    m = mn;
  }
  // Lanes L and L^16 hold the same query column with disjoint key rows.
  const float mo = __shfl_xor(m, 16, 32);
  const float Zo = __shfl_xor(Z, 16, 32);
  const float mc = fmaxf(m, mo);
  const float Zc = Z * __expf(m - mc) + Zo * __expf(mo - mc);
  if (lane < 16) {
    v2f st; st.x = mc; st.y = Zc;
    stats[(size_t)bh * 1024 + qt * 16 + nlo] = st;
  }
}

// ---------------------------------------------------------------------------
// Kernel 2b: s[key] = sum_q exp(score - m_q)/Z_q.  A = Q-tile (queries as
// rows, looped), B = K-tile (keys as cols, loaded once).
// ---------------------------------------------------------------------------
__global__ __launch_bounds__(256) void key_weight_kernel(
    const _Float16* __restrict__ Qh, const _Float16* __restrict__ Kh,
    const v2f* __restrict__ stats, float* __restrict__ Ssum)
{
  const int bh = blockIdx.x;
  const int b = bh >> 3, h = bh & 7;
  const int wave = threadIdx.x >> 5, lane = threadIdx.x & 31;
  const int nlo = lane & 15, hf = lane >> 4;
  const int ktile = blockIdx.y * 8 + wave;         // key tile 0..63
  const float scale = 0.17677669529663687f;

  const size_t krow = ((size_t)b * 1024 + ktile * 16 + nlo) * EDIM + h * 32;
  V16 bf;
  bf.h2[0] = *(const v8h*)(Kh + krow + hf * 8);
  bf.h2[1] = *(const v8h*)(Kh + krow + 16 + hf * 8);

  float acc = 0.f;
  for (int qt = 0; qt < 64; ++qt) {
    const size_t qrow = ((size_t)b * 1024 + qt * 16 + nlo) * EDIM + h * 32;
    V16 af;
    af.h2[0] = *(const v8h*)(Qh + qrow + hf * 8);
    af.h2[1] = *(const v8h*)(Qh + qrow + 16 + hf * 8);
    v8f c = vzero8();
    c = WMMA_F16(af.v, bf.v, c);
    // C rows (queries) per lane: qt*16 + hf*8 + r.
    const v2f* stp = stats + (size_t)bh * 1024 + qt * 16 + hf * 8;
    #pragma unroll
    for (int r = 0; r < 8; ++r) {
      const v2f st = stp[r];
      acc += __expf(c[r] * scale - st.x) * __builtin_amdgcn_rcpf(st.y);
    }
  }
  acc += __shfl_xor(acc, 16, 32);
  if (lane < 16) Ssum[(size_t)bh * 1024 + ktile * 16 + nlo] = acc;
}

// ---------------------------------------------------------------------------
// Kernel 3: out = (V * s) @ Wo + bo, f32 output.  Same structure as kernel 1;
// each 32-wide k-chunk lies inside a single head, so s is one scalar per
// (lane-row, k-step).
// ---------------------------------------------------------------------------
__global__ __launch_bounds__(256) void out_proj_kernel(
    const _Float16* __restrict__ Vh, const float* __restrict__ Wo,
    const float* __restrict__ bo, const float* __restrict__ Ssum,
    float* __restrict__ Y)
{
  __shared__ _Float16 Wt[EDIM * LDSW];
  const int tid   = threadIdx.x;
  const int wave  = tid >> 5;
  const int lane  = tid & 31;
  const int nlo   = lane & 15;
  const int hf    = lane >> 4;
  const int mbase = blockIdx.x * 128 + wave * 16;
  const int mrow  = mbase + nlo;
  const int b  = mrow >> 10;
  const int tf = mrow & 1023;

  v8f acc[16];
  #pragma unroll
  for (int t = 0; t < 16; ++t) acc[t] = vzero8();

  for (int k0 = 0; k0 < EDIM; k0 += 32) {
    __syncthreads();
    #pragma unroll
    for (int it = 0; it < 8; ++it) {
      const int i  = tid + it * 256;
      const int kk = i >> 6;
      const int n4 = (i & 63) << 2;
      const v4f w4 = *(const v4f*)(Wo + (size_t)(k0 + kk) * EDIM + n4);
      Wt[(n4 + 0) * LDSW + kk] = (_Float16)w4.x;
      Wt[(n4 + 1) * LDSW + kk] = (_Float16)w4.y;
      Wt[(n4 + 2) * LDSW + kk] = (_Float16)w4.z;
      Wt[(n4 + 3) * LDSW + kk] = (_Float16)w4.w;
    }
    __syncthreads();

    const int hh = k0 >> 5;                               // head for this k-chunk
    const float sv = Ssum[((size_t)b * 8 + hh) * 1024 + tf];
    const _Float16* vp = Vh + (size_t)mrow * EDIM + k0 + hf * 8;
    const v8h x0 = *(const v8h*)(vp);
    const v8h x1 = *(const v8h*)(vp + 16);
    V16 af;
    #pragma unroll
    for (int j = 0; j < 8; ++j) {
      af.v[j]     = (_Float16)((float)x0[j] * sv);
      af.v[j + 8] = (_Float16)((float)x1[j] * sv);
    }
    #pragma unroll
    for (int nt = 0; nt < 16; ++nt) {
      const _Float16* bp = &Wt[(nt * 16 + nlo) * LDSW + hf * 8];
      V16 bf;
      bf.h2[0] = *(const v8h*)(bp);
      bf.h2[1] = *(const v8h*)(bp + 16);
      acc[nt] = WMMA_F16(af.v, bf.v, acc[nt]);
    }
  }

  #pragma unroll
  for (int nt = 0; nt < 16; ++nt) {
    const int n = nt * 16 + nlo;
    const float bb = bo[n];
    #pragma unroll
    for (int r = 0; r < 8; ++r) {
      const int m = mbase + r + hf * 8;
      Y[(size_t)m * EDIM + n] = acc[nt][r] + bb;
    }
  }
}

// ---------------------------------------------------------------------------
extern "C" void kernel_launch(void* const* d_in, const int* in_sizes, int n_in,
                              void* d_out, int out_size, void* d_ws, size_t ws_size,
                              hipStream_t stream) {
  (void)in_sizes; (void)n_in; (void)out_size; (void)ws_size;
  const float* q  = (const float*)d_in[0];
  const float* k  = (const float*)d_in[1];
  const float* v  = (const float*)d_in[2];
  const float* Wq = (const float*)d_in[3];
  const float* bq = (const float*)d_in[4];
  const float* Wk = (const float*)d_in[5];
  const float* bk = (const float*)d_in[6];
  const float* Wv = (const float*)d_in[7];
  const float* bv = (const float*)d_in[8];
  const float* Wo = (const float*)d_in[9];
  const float* bo = (const float*)d_in[10];

  // Workspace layout: Qh/Kh/Vh f16 (64 MiB each) + stats (512 KiB) + s (256 KiB)
  char* ws = (char*)d_ws;
  const size_t halfBytes = (size_t)MTOT * EDIM * sizeof(_Float16);
  _Float16* Qh   = (_Float16*)(ws);
  _Float16* Kh   = (_Float16*)(ws + halfBytes);
  _Float16* Vh   = (_Float16*)(ws + 2 * halfBytes);
  v2f*      stat = (v2f*)    (ws + 3 * halfBytes);
  float*    Ssum = (float*)  (ws + 3 * halfBytes + (size_t)64 * 1024 * sizeof(v2f));

  proj_f16_kernel<<<1024, 256, 0, stream>>>(q, Wq, bq, Qh);
  proj_f16_kernel<<<1024, 256, 0, stream>>>(k, Wk, bk, Kh);
  proj_f16_kernel<<<1024, 256, 0, stream>>>(v, Wv, bv, Vh);
  softmax_stats_kernel<<<dim3(64, 8), 256, 0, stream>>>(Qh, Kh, stat);
  key_weight_kernel<<<dim3(64, 8), 256, 0, stream>>>(Qh, Kh, stat, Ssum);
  out_proj_kernel<<<1024, 256, 0, stream>>>(Vh, Wo, bo, Ssum, (float*)d_out);
}
